// SimResultPredictionModel_7310034338069
// MI455X (gfx1250) — matrix operand order repaired and checked by reading, not compiled, not run on hardware
//
#include <hip/hip_runtime.h>

typedef __attribute__((ext_vector_type(16))) __bf16 v16bf;
typedef __attribute__((ext_vector_type(8)))  float  v8f;
typedef __attribute__((ext_vector_type(4)))  unsigned int v4u;
typedef __attribute__((ext_vector_type(8)))  int v8i;
typedef __attribute__((ext_vector_type(4)))  int v4i;

#if defined(__has_builtin)
#  if __has_builtin(__builtin_amdgcn_tensor_load_to_lds)
#    define HAVE_TDM 1
#  endif
#endif
#ifndef HAVE_TDM
#  define HAVE_TDM 0
#endif

struct Q2 { uint4 a, b; };   // 32 bytes == one v16bf fragment

static constexpr float BNS = 0.99999500003749969f;  // 1/sqrt(1+1e-5)

__device__ inline unsigned short f32_to_bf16(float f) {
  unsigned int u = __float_as_uint(f);
  unsigned int lsb = (u >> 16) & 1u;
  u += 0x7fffu + lsb;
  return (unsigned short)(u >> 16);
}
__device__ inline float bf16_to_f32(unsigned short h) {
  return __uint_as_float(((unsigned int)h) << 16);
}
__device__ inline float gelu_exact(float x) {
  return 0.5f * x * (1.0f + erff(x * 0.70710678118654752f));
}

// LDS aperture: generic flat address low 32 bits == wave-relative LDS offset.
__device__ inline unsigned lds_addr_of(const void* p) {
  return (unsigned)(unsigned long long)p;
}

// Async global->LDS copy, 16B per active lane (GV mode, ASYNCcnt-tracked).
__device__ inline void async_copy16(unsigned lds_off, const void* __restrict__ g) {
  asm volatile("global_load_async_to_lds_b128 %0, %1, off"
               :: "v"(lds_off), "v"(g)
               : "memory");
}
__device__ inline void wait_async0() {
  asm volatile("s_wait_asynccnt 0" ::: "memory");
}

// TDM: stage `bytes/4` contiguous f32 (one 1-row 2D tile) into LDS.
// D# per ISA 8.3-8.4: count=1, type=2, data_size=4B, tile = nelem x 1.
__device__ inline void tdm_stage_f32(unsigned lds_off, const float* gsrc, int nelem) {
#if HAVE_TDM
  unsigned long long ga = (unsigned long long)(const void*)gsrc;
  v4u g0;
  g0[0] = 1u;                                                 // count=1
  g0[1] = lds_off;                                            // lds_addr
  g0[2] = (unsigned)(ga & 0xFFFFFFFFu);                       // global_addr lo
  g0[3] = (unsigned)((ga >> 32) & 0x1FFFFFFu) | 0x80000000u;  // addr[56:32] | type=2
  v8i g1;
  g1[0] = 0x20000;                        // data_size code 2 (4B) at bits[17:16]
  g1[1] = (int)((unsigned)nelem << 16);   // tensor_dim0[15:0]
  g1[2] = (int)(1u << 16);                // tensor_dim0 hi=0 | tensor_dim1=1
  g1[3] = (int)((unsigned)nelem << 16);   // tensor_dim1 hi=0 | tile_dim0=nelem
  g1[4] = 1;                              // tile_dim1=1, tile_dim2=0
  g1[5] = nelem;                          // tensor_dim0_stride lo
  g1[6] = 0;
  g1[7] = 0;
  v4i gz = {0, 0, 0, 0};
#if __clang_major__ >= 23
  v8i gz8 = {0, 0, 0, 0, 0, 0, 0, 0};
  __builtin_amdgcn_tensor_load_to_lds(g0, g1, gz, gz, gz8, 0);
#else
  __builtin_amdgcn_tensor_load_to_lds(g0, g1, gz, gz, 0);
#endif
#endif
}

// A fragment (16x32 bf16, ISA 7.12.2 layout) from LDS.
__device__ inline v16bf load_frag_lds(const unsigned short* p, int stride,
                                      int rowbase, int kbase, int lane) {
  int r = rowbase + (lane & 15);
  int k = kbase + ((lane >> 4) << 3);
  Q2 q;
  q.a = *reinterpret_cast<const uint4*>(p + r * stride + k);
  q.b = *reinterpret_cast<const uint4*>(p + r * stride + k + 16);
  return __builtin_bit_cast(v16bf, q);
}

// B fragment from pre-swizzled global weights: block (kt, nt) is 32 lanes x 32B.
__device__ inline v16bf load_frag_gbl(const unsigned short* __restrict__ bp,
                                      int kt, int nt, int lane) {
  const uint4* p = reinterpret_cast<const uint4*>(
      bp + ((((kt << 5) + nt) << 5) + lane) * 16);
  Q2 q; q.a = p[0]; q.b = p[1];
  return __builtin_bit_cast(v16bf, q);
}

__device__ inline v8f wmma_bf16(v16bf a, v16bf b, v8f c) {
  return __builtin_amdgcn_wmma_f32_16x16x32_bf16(false, a, false, b,
                                                 (short)0, c, false, false);
}

// ---------------------------------------------------------------------------
__global__ void __launch_bounds__(256)
pack_b_kernel(const float* __restrict__ W, int Krows, int KT,
              unsigned int* __restrict__ out) {
  int idx = blockIdx.x * 256 + threadIdx.x;
  int total = KT * 8192;
  if (idx >= total) return;
  int i    = idx & 7;
  int lane = (idx >> 3) & 31;
  int nt   = (idx >> 8) & 31;
  int kt   = idx >> 13;
  int col  = nt * 16 + (lane & 15);
  int k0   = kt * 32 + ((lane >> 4) << 3) + 2 * i + ((i >> 2) << 3);
  float w0 = (k0     < Krows) ? W[k0 * 512 + col]       : 0.f;
  float w1 = (k0 + 1 < Krows) ? W[(k0 + 1) * 512 + col] : 0.f;
  out[idx] = (unsigned int)f32_to_bf16(w0) |
             ((unsigned int)f32_to_bf16(w1) << 16);
}

__global__ void __launch_bounds__(256)
init_gacc_kernel(const float* __restrict__ gf, float* __restrict__ gacc, int n) {
  int i = blockIdx.x * 256 + threadIdx.x;
  if (i < n) gacc[i] = gf[i];
}

// ---------------------------------------------------------------------------
// Node path. 32 rows/WG, 8 waves. Wave w owns N-tiles w*4..w*4+3 and both
// M-tiles (each B frag feeds 2 WMMAs, no cross-wave B duplication).
__global__ void __launch_bounds__(256)
node_kernel(const float* __restrict__ nf, const float* __restrict__ dp,
            const int* __restrict__ node_y, const int* __restrict__ batch_idx,
            const unsigned short* __restrict__ w1p, const float* __restrict__ b1,
            const float* __restrict__ g1, const float* __restrict__ be1,
            const unsigned short* __restrict__ w2p, const float* __restrict__ b2,
            float* __restrict__ gacc, int N) {
  constexpr int KAS = 72;   // 64 + 8 pad (bf16 elems), 144B rows -> 16B aligned
  constexpr int HS  = 520;  // 512 + 8 pad
  __shared__ __align__(16) float          Sf[32 * 32];   // raw nf stage (4KB)
  __shared__ __align__(16) unsigned short Xt[32 * KAS];
  __shared__ __align__(16) unsigned short Hb[32 * HS];

  const int t = threadIdx.x;
  const int rowbase = blockIdx.x * 32;

  // --- stage the 4KB nf tile into LDS via the Tensor Data Mover ---
  if (rowbase + 32 <= N) {
#if HAVE_TDM
    if (t == 0)
      tdm_stage_f32(lds_addr_of(Sf), nf + (size_t)rowbase * 32, 32 * 32);
    __builtin_amdgcn_s_wait_tensorcnt((short)0);
#else
    async_copy16(lds_addr_of(Sf) + t * 16,
                 (const char*)(nf + (size_t)rowbase * 32) + t * 16);
    wait_async0();
#endif
  } else {
    for (int e = t; e < 32 * 32; e += 256) {
      int r = e >> 5, c = e & 31;
      int grow = rowbase + r;
      Sf[e] = (grow < N) ? nf[grow * 32 + c] : 0.f;
    }
  }
  __syncthreads();

  // build augmented bf16 input tile [32 x 64] (cols 41..63 zero)
  for (int e = t; e < 32 * 64; e += 256) {
    int r = e >> 6, c = e & 63;
    int grow = rowbase + r;
    float v = 0.f;
    if (c < 32)                  v = Sf[r * 32 + c];
    else if (c < 41 && grow < N) v = (node_y[grow] == (c - 32)) ? dp[grow] : 0.f;
    Xt[r * KAS + c] = f32_to_bf16(v);
  }
  __syncthreads();

  const int w = t >> 5, lane = t & 31;
  const int lcol = lane & 15, lhalf = lane >> 4;
  const v8f zero = {0.f, 0.f, 0.f, 0.f, 0.f, 0.f, 0.f, 0.f};

  // ---- GEMM1: [32x64] @ [64x512] ----
  v8f acc[2][4];
#pragma unroll
  for (int m = 0; m < 2; ++m)
#pragma unroll
    for (int j = 0; j < 4; ++j) acc[m][j] = zero;
#pragma unroll
  for (int kt = 0; kt < 2; ++kt) {
    v16bf a0 = load_frag_lds(Xt, KAS, 0,  kt * 32, lane);
    v16bf a1 = load_frag_lds(Xt, KAS, 16, kt * 32, lane);
#pragma unroll
    for (int j = 0; j < 4; ++j) {
      v16bf b = load_frag_gbl(w1p, kt, w * 4 + j, lane);
      acc[0][j] = wmma_bf16(a0, b, acc[0][j]);
      acc[1][j] = wmma_bf16(a1, b, acc[1][j]);
    }
  }

  // epilogue: bias + BN + GELU -> bf16 hidden in LDS
#pragma unroll
  for (int j = 0; j < 4; ++j) {
    int col = (w * 4 + j) * 16 + lcol;
    float bb = b1[col];
    float sc = g1[col] * BNS;
    float bt = be1[col];
#pragma unroll
    for (int m = 0; m < 2; ++m)
#pragma unroll
      for (int v = 0; v < 8; ++v) {
        float x = acc[m][j][v] + bb;
        x = x * sc + bt;
        x = gelu_exact(x);
        Hb[(m * 16 + (lhalf << 3) + v) * HS + col] = f32_to_bf16(x);
      }
  }
  __syncthreads();

  // ---- GEMM2: [32x512] @ [512x512] ----
  v8f acc2[2][4];
#pragma unroll
  for (int m = 0; m < 2; ++m)
#pragma unroll
    for (int j = 0; j < 4; ++j) acc2[m][j] = zero;
  for (int kt = 0; kt < 16; ++kt) {
    __builtin_prefetch(w2p + (((((kt + 1) & 15) * 32 + w * 4) * 32 + lane) * 16), 0, 0);
    v16bf a0 = load_frag_lds(Hb, HS, 0,  kt * 32, lane);
    v16bf a1 = load_frag_lds(Hb, HS, 16, kt * 32, lane);
#pragma unroll
    for (int j = 0; j < 4; ++j) {
      v16bf b = load_frag_gbl(w2p, kt, w * 4 + j, lane);
      acc2[0][j] = wmma_bf16(a0, b, acc2[0][j]);
      acc2[1][j] = wmma_bf16(a1, b, acc2[1][j]);
    }
  }

  // bias + atomic scatter into graph accumulator (L2-resident)
#pragma unroll
  for (int j = 0; j < 4; ++j) {
    int col = (w * 4 + j) * 16 + lcol;
    float bb = b2[col];
#pragma unroll
    for (int m = 0; m < 2; ++m)
#pragma unroll
      for (int v = 0; v < 8; ++v) {
        int grow = rowbase + m * 16 + (lhalf << 3) + v;
        if (grow < N) {
          int g = batch_idx[grow];
          atomicAdd(&gacc[g * 512 + col], acc2[m][j][v] + bb);
        }
      }
  }
}

// ---------------------------------------------------------------------------
// Edge path: ef -> MLP -> scatter by batch_idx[edge_index[0][e]].
// Input tile staged with ASYNCcnt global->LDS copies.
__global__ void __launch_bounds__(256)
edge_kernel(const float* __restrict__ ef, const int* __restrict__ ei,
            const int* __restrict__ batch_idx,
            const unsigned short* __restrict__ w1p, const float* __restrict__ b1,
            const float* __restrict__ g1, const float* __restrict__ be1,
            const unsigned short* __restrict__ w2p, const float* __restrict__ b2,
            float* __restrict__ gacc, int E) {
  constexpr int KAS = 40;   // 32 + 8 pad -> 80B rows, 16B aligned
  constexpr int HS  = 520;
  __shared__ __align__(16) float          Sf[32 * 16];   // raw ef stage (2KB)
  __shared__ __align__(16) unsigned short Xt[32 * KAS];
  __shared__ __align__(16) unsigned short Hb[32 * HS];

  const int t = threadIdx.x;
  const int rowbase = blockIdx.x * 32;

  if (rowbase + 32 <= E) {
    if (t < 128) {
      const char* src = (const char*)(ef + (size_t)rowbase * 16);
      async_copy16(lds_addr_of(Sf) + t * 16, src + t * 16);
    }
    wait_async0();
  } else {
    for (int e = t; e < 32 * 16; e += 256) {
      int r = e >> 4, c = e & 15;
      int grow = rowbase + r;
      Sf[e] = (grow < E) ? ef[grow * 16 + c] : 0.f;
    }
  }
  __syncthreads();

  for (int e = t; e < 32 * 32; e += 256) {
    int r = e >> 5, c = e & 31;
    float v = (c < 16) ? Sf[r * 16 + c] : 0.f;
    Xt[r * KAS + c] = f32_to_bf16(v);
  }
  __syncthreads();

  const int w = t >> 5, lane = t & 31;
  const int lcol = lane & 15, lhalf = lane >> 4;
  const v8f zero = {0.f, 0.f, 0.f, 0.f, 0.f, 0.f, 0.f, 0.f};

  v8f acc[2][4];
#pragma unroll
  for (int m = 0; m < 2; ++m)
#pragma unroll
    for (int j = 0; j < 4; ++j) acc[m][j] = zero;
  {
    v16bf a0 = load_frag_lds(Xt, KAS, 0,  0, lane);
    v16bf a1 = load_frag_lds(Xt, KAS, 16, 0, lane);
#pragma unroll
    for (int j = 0; j < 4; ++j) {
      v16bf b = load_frag_gbl(w1p, 0, w * 4 + j, lane);
      acc[0][j] = wmma_bf16(a0, b, acc[0][j]);
      acc[1][j] = wmma_bf16(a1, b, acc[1][j]);
    }
  }

#pragma unroll
  for (int j = 0; j < 4; ++j) {
    int col = (w * 4 + j) * 16 + lcol;
    float bb = b1[col];
    float sc = g1[col] * BNS;
    float bt = be1[col];
#pragma unroll
    for (int m = 0; m < 2; ++m)
#pragma unroll
      for (int v = 0; v < 8; ++v) {
        float x = acc[m][j][v] + bb;
        x = x * sc + bt;
        x = gelu_exact(x);
        Hb[(m * 16 + (lhalf << 3) + v) * HS + col] = f32_to_bf16(x);
      }
  }
  __syncthreads();

  v8f acc2[2][4];
#pragma unroll
  for (int m = 0; m < 2; ++m)
#pragma unroll
    for (int j = 0; j < 4; ++j) acc2[m][j] = zero;
  for (int kt = 0; kt < 16; ++kt) {
    __builtin_prefetch(w2p + (((((kt + 1) & 15) * 32 + w * 4) * 32 + lane) * 16), 0, 0);
    v16bf a0 = load_frag_lds(Hb, HS, 0,  kt * 32, lane);
    v16bf a1 = load_frag_lds(Hb, HS, 16, kt * 32, lane);
#pragma unroll
    for (int j = 0; j < 4; ++j) {
      v16bf b = load_frag_gbl(w2p, kt, w * 4 + j, lane);
      acc2[0][j] = wmma_bf16(a0, b, acc2[0][j]);
      acc2[1][j] = wmma_bf16(a1, b, acc2[1][j]);
    }
  }

#pragma unroll
  for (int j = 0; j < 4; ++j) {
    int col = (w * 4 + j) * 16 + lcol;
    float bb = b2[col];
#pragma unroll
    for (int m = 0; m < 2; ++m)
#pragma unroll
      for (int v = 0; v < 8; ++v) {
        int grow = rowbase + m * 16 + (lhalf << 3) + v;
        if (grow < E) {
          int g = batch_idx[ei[grow]];   // ei row 0 = source node ids
          atomicAdd(&gacc[g * 512 + col], acc2[m][j][v] + bb);
        }
      }
  }
}

// ---------------------------------------------------------------------------
// Head: two 512x512 hidden layers (WMMA) + scalar [512x2] output.
__global__ void __launch_bounds__(256)
head_kernel(const float* __restrict__ gacc,
            const unsigned short* __restrict__ w1p, const float* __restrict__ b1,
            const float* __restrict__ g1, const float* __restrict__ be1,
            const unsigned short* __restrict__ w2p, const float* __restrict__ b2,
            const float* __restrict__ g2, const float* __restrict__ be2,
            const float* __restrict__ w3, const float* __restrict__ b3,
            float* __restrict__ out, int G) {
  constexpr int HS = 520;
  __shared__ __align__(16) unsigned short A[32 * HS];
  const int t = threadIdx.x;
  const int rowbase = blockIdx.x * 32;

  for (int e = t; e < 32 * 512; e += 256) {
    int r = e >> 9, c = e & 511;
    float v = (rowbase + r < G) ? gacc[(rowbase + r) * 512 + c] : 0.f;
    A[r * HS + c] = f32_to_bf16(v);
  }
  __syncthreads();

  const int w = t >> 5, lane = t & 31;
  const int lcol = lane & 15, lhalf = lane >> 4;
  const v8f zero = {0.f, 0.f, 0.f, 0.f, 0.f, 0.f, 0.f, 0.f};

  for (int layer = 0; layer < 2; ++layer) {
    const unsigned short* wp = (layer == 0) ? w1p : w2p;
    const float* bb_ = (layer == 0) ? b1 : b2;
    const float* gg_ = (layer == 0) ? g1 : g2;
    const float* be_ = (layer == 0) ? be1 : be2;

    v8f acc[2][4];
#pragma unroll
    for (int m = 0; m < 2; ++m)
#pragma unroll
      for (int j = 0; j < 4; ++j) acc[m][j] = zero;
    for (int kt = 0; kt < 16; ++kt) {
      v16bf a0 = load_frag_lds(A, HS, 0,  kt * 32, lane);
      v16bf a1 = load_frag_lds(A, HS, 16, kt * 32, lane);
#pragma unroll
      for (int j = 0; j < 4; ++j) {
        v16bf b = load_frag_gbl(wp, kt, w * 4 + j, lane);
        acc[0][j] = wmma_bf16(a0, b, acc[0][j]);
        acc[1][j] = wmma_bf16(a1, b, acc[1][j]);
      }
    }
    __syncthreads();   // all reads of A done before overwriting
#pragma unroll
    for (int j = 0; j < 4; ++j) {
      int col = (w * 4 + j) * 16 + lcol;
      float bb = bb_[col];
      float sc = gg_[col] * BNS;
      float bt = be_[col];
#pragma unroll
      for (int m = 0; m < 2; ++m)
#pragma unroll
        for (int v = 0; v < 8; ++v) {
          float x = acc[m][j][v] + bb;
          x = x * sc + bt;
          x = gelu_exact(x);
          A[(m * 16 + (lhalf << 3) + v) * HS + col] = f32_to_bf16(x);
        }
    }
    __syncthreads();
  }

  if (t < 64) {
    int r = t >> 1, c = t & 1;
    float s = 0.f;
    for (int k = 0; k < 512; ++k)
      s += bf16_to_f32(A[r * HS + k]) * w3[k * 2 + c];
    if (rowbase + r < G) out[(rowbase + r) * 2 + c] = s + b3[c];
  }
}

// ---------------------------------------------------------------------------
extern "C" void kernel_launch(void* const* d_in, const int* in_sizes, int n_in,
                              void* d_out, int out_size, void* d_ws, size_t ws_size,
                              hipStream_t stream) {
  const float* nf        = (const float*)d_in[0];
  const float* ef        = (const float*)d_in[1];
  const float* gf        = (const float*)d_in[2];
  const float* dp        = (const float*)d_in[3];
  const int*   node_y    = (const int*)d_in[4];
  const int*   batch_idx = (const int*)d_in[5];
  const int*   ei        = (const int*)d_in[6];
  const float* nf_w1     = (const float*)d_in[7];
  const float* nf_b1     = (const float*)d_in[8];
  const float* nf_g1     = (const float*)d_in[9];
  const float* nf_be1    = (const float*)d_in[10];
  const float* nf_w2     = (const float*)d_in[11];
  const float* nf_b2     = (const float*)d_in[12];
  const float* ef_w1     = (const float*)d_in[13];
  const float* ef_b1     = (const float*)d_in[14];
  const float* ef_g1     = (const float*)d_in[15];
  const float* ef_be1    = (const float*)d_in[16];
  const float* ef_w2     = (const float*)d_in[17];
  const float* ef_b2     = (const float*)d_in[18];
  const float* ow1       = (const float*)d_in[19];
  const float* ob1       = (const float*)d_in[20];
  const float* og1       = (const float*)d_in[21];
  const float* obe1      = (const float*)d_in[22];
  const float* ow2       = (const float*)d_in[23];
  const float* ob2       = (const float*)d_in[24];
  const float* og2       = (const float*)d_in[25];
  const float* obe2      = (const float*)d_in[26];
  const float* ow3       = (const float*)d_in[27];
  const float* ob3       = (const float*)d_in[28];

  const int N = in_sizes[3];        // device_params
  const int E = in_sizes[1] / 16;   // ef
  const int G = in_sizes[2] / 512;  // gf

  // workspace carve-up
  char* ws = (char*)d_ws;
  size_t off = 0;
  auto carve = [&](size_t bytes) {
    off = (off + 255) & ~(size_t)255;
    void* p = ws + off;
    off += bytes;
    return p;
  };
  float*          gacc = (float*)carve((size_t)G * 512 * 4);
  unsigned short* w1n  = (unsigned short*)carve((size_t)2  * 8192 * 4);
  unsigned short* w2n  = (unsigned short*)carve((size_t)16 * 8192 * 4);
  unsigned short* w1e  = (unsigned short*)carve((size_t)1  * 8192 * 4);
  unsigned short* w2e  = (unsigned short*)carve((size_t)16 * 8192 * 4);
  unsigned short* ow1p = (unsigned short*)carve((size_t)16 * 8192 * 4);
  unsigned short* ow2p = (unsigned short*)carve((size_t)16 * 8192 * 4);

  // pack weights to bf16 B-fragment layout (per launch; deterministic)
  pack_b_kernel<<<(2  * 8192 + 255) / 256, 256, 0, stream>>>(nf_w1, 41,  2,  (unsigned int*)w1n);
  pack_b_kernel<<<(16 * 8192 + 255) / 256, 256, 0, stream>>>(nf_w2, 512, 16, (unsigned int*)w2n);
  pack_b_kernel<<<(1  * 8192 + 255) / 256, 256, 0, stream>>>(ef_w1, 16,  1,  (unsigned int*)w1e);
  pack_b_kernel<<<(16 * 8192 + 255) / 256, 256, 0, stream>>>(ef_w2, 512, 16, (unsigned int*)w2e);
  pack_b_kernel<<<(16 * 8192 + 255) / 256, 256, 0, stream>>>(ow1,   512, 16, (unsigned int*)ow1p);
  pack_b_kernel<<<(16 * 8192 + 255) / 256, 256, 0, stream>>>(ow2,   512, 16, (unsigned int*)ow2p);

  init_gacc_kernel<<<(G * 512 + 255) / 256, 256, 0, stream>>>(gf, gacc, G * 512);

  node_kernel<<<(N + 31) / 32, 256, 0, stream>>>(
      nf, dp, node_y, batch_idx, w1n, nf_b1, nf_g1, nf_be1, w2n, nf_b2, gacc, N);

  edge_kernel<<<(E + 31) / 32, 256, 0, stream>>>(
      ef, ei, batch_idx, w1e, ef_b1, ef_g1, ef_be1, w2e, ef_b2, gacc, E);

  head_kernel<<<(G + 31) / 32, 256, 0, stream>>>(
      gacc, ow1p, ob1, og1, obe1, ow2p, ob2, og2, obe2, ow3, ob3,
      (float*)d_out, G);
}
